// Decoder_59072980189740
// MI455X (gfx1250) — matrix operand order repaired and checked by reading, not compiled, not run on hardware
//
#include <hip/hip_runtime.h>

// ---------------------------------------------------------------- constants
#define BATCH  256
#define LATENT 512
#define HID    1024
#define OUTD   256
#define SEQLEN 128

// ---------------------------------------------------------------- types
typedef unsigned short bfbits;                                    // raw bf16 bits
typedef __attribute__((ext_vector_type(16))) __bf16 v16bf;        // WMMA A/B frag
typedef __attribute__((ext_vector_type(8)))  float  v8f;          // WMMA C/D frag

__device__ __forceinline__ v8f vzero8() {
    v8f z;
#pragma unroll
    for (int i = 0; i < 8; ++i) z[i] = 0.0f;
    return z;
}

// f32 -> bf16 bits, round-to-nearest-even
__device__ __forceinline__ bfbits f2bf(float f) {
    unsigned int u = __float_as_uint(f);
    unsigned int lsb = (u >> 16) & 1u;
    u += 0x7fffu + lsb;
    return (bfbits)(u >> 16);
}

__device__ __forceinline__ float sigmoidf_(float x) {
    return 1.0f / (1.0f + __expf(-x));
}

// ---------------------------------------------------------------- WMMA fragment loads
// A fragment: 16x32 bf16 tile, X row-major [*, ldx]; rows row0..row0+15, cols k0..k0+31.
// Layout (ISA 7.12.2): lane l -> M = l&15; K chunks {8*(l>>4) .. +7} and {+16 .. +23}.
__device__ __forceinline__ v16bf load_a_frag(const bfbits* __restrict__ X, int ldx,
                                             int row0, int k0, int lane) {
    int m  = lane & 15;
    int kh = (lane >> 4) << 3;            // 0 or 8
    const bfbits* p = X + (size_t)(row0 + m) * ldx + k0 + kh;
    union { v16bf v; uint4 q[2]; } f;
    f.q[0] = *reinterpret_cast<const uint4*>(p);        // K = kh .. kh+7
    f.q[1] = *reinterpret_cast<const uint4*>(p + 16);   // K = kh+16 .. kh+23
    return f.v;
}

// B fragment: 32x16 bf16 (KxN) where B[k,n] = W[n0+n, k], W row-major [N, ldw].
// Layout: lane l -> N = l&15; K chunks {8*(l>>4) .. +7} and {+16 .. +23} (matches A).
__device__ __forceinline__ v16bf load_b_frag(const bfbits* __restrict__ W, int ldw,
                                             int n0, int k0, int lane) {
    int n  = lane & 15;
    int kh = (lane >> 4) << 3;            // 0 or 8
    const bfbits* p = W + (size_t)(n0 + n) * ldw + k0 + kh;
    union { v16bf v; uint4 q[2]; } f;
    f.q[0] = *reinterpret_cast<const uint4*>(p);
    f.q[1] = *reinterpret_cast<const uint4*>(p + 16);
    return f.v;
}

#define WMMA_BF16(A, Bf, C) \
    __builtin_amdgcn_wmma_f32_16x16x32_bf16(false, (A), false, (Bf), (short)0, (C), false, false)

// ---------------------------------------------------------------- kernel 1: convert / init
__global__ __launch_bounds__(256) void convert_kernel(
    const float* __restrict__ latent, const float* __restrict__ fc_W,
    const float* __restrict__ W_ih,   const float* __restrict__ W_hh,
    const float* __restrict__ b_ih,   const float* __restrict__ b_hh,
    const float* __restrict__ out_W,  const float* __restrict__ out_b,
    bfbits* __restrict__ lat_bf,  bfbits* __restrict__ fcW_bf,
    bfbits* __restrict__ Wih_bf,  bfbits* __restrict__ Whh_bf,
    bfbits* __restrict__ outW_bf, bfbits* __restrict__ outWT_bf,
    float* __restrict__ bias0, float* __restrict__ bias_eff, float* __restrict__ cstate)
{
    int tid = blockIdx.x * blockDim.x + threadIdx.x;
    int stride = gridDim.x * blockDim.x;

    for (int i = tid; i < BATCH * LATENT; i += stride) lat_bf[i] = f2bf(latent[i]);
    for (int i = tid; i < HID * LATENT;   i += stride) fcW_bf[i] = f2bf(fc_W[i]);
    for (int i = tid; i < 4 * HID * OUTD; i += stride) Wih_bf[i] = f2bf(W_ih[i]);
    for (int i = tid; i < 4 * HID * HID;  i += stride) Whh_bf[i] = f2bf(W_hh[i]);
    for (int i = tid; i < OUTD * HID; i += stride) {
        float v = out_W[i];
        outW_bf[i] = f2bf(v);
        int r = i >> 10;          // row in [0,256)   (HID = 1024 cols)
        int c = i & (HID - 1);    // col in [0,1024)
        outWT_bf[(size_t)c * OUTD + r] = f2bf(v);     // transposed copy [HID, OUTD]
    }
    for (int i = tid; i < BATCH * HID; i += stride) cstate[i] = 0.0f;
    for (int j = tid; j < 4 * HID; j += stride) {
        float b0 = b_ih[j] + b_hh[j];
        bias0[j] = b0;
        float s = 0.0f;
        for (int m = 0; m < OUTD; ++m) s += W_ih[(size_t)j * OUTD + m] * out_b[m];
        bias_eff[j] = b0 + s;                         // fold W_ih @ out_b
    }
}

// ---------------------------------------------------------------- kernel 2: W_eff = W_hh + W_ih @ out_W
// M = 4*HID (rows j), N = HID (cols k), K = OUTD. One wave per 16x16 tile.
__global__ __launch_bounds__(256) void weff_kernel(
    const bfbits* __restrict__ Wih_bf, const bfbits* __restrict__ outWT_bf,
    const float* __restrict__ W_hh, bfbits* __restrict__ Weff)
{
    int wave = blockIdx.x * (blockDim.x >> 5) + (threadIdx.x >> 5);
    int lane = threadIdx.x & 31;
    if (wave >= (4 * HID / 16) * (HID / 16)) return;
    int jt = wave >> 6;                 // 256 tiles over 4*HID
    int kt = wave & 63;                 // 64 tiles over HID
    int j0 = jt * 16, k0 = kt * 16;

    v8f acc = vzero8();
    for (int m = 0; m < OUTD; m += 32) {
        v16bf a = load_a_frag(Wih_bf, OUTD, j0, m, lane);
        v16bf b = load_b_frag(outWT_bf, OUTD, k0, m, lane);   // B[m,k] = out_W[m, k0+n]
        acc = WMMA_BF16(a, b, acc);
    }
    int n = lane & 15, mh = (lane >> 4) << 3;
#pragma unroll
    for (int v = 0; v < 8; ++v) {
        int j = j0 + mh + v, k = k0 + n;
        Weff[(size_t)j * HID + k] = f2bf(acc[v] + W_hh[(size_t)j * HID + k]);
    }
}

// ---------------------------------------------------------------- kernel 3: h0 = latent @ fc_W^T + fc_b
__global__ __launch_bounds__(256) void h0_kernel(
    const bfbits* __restrict__ lat_bf, const bfbits* __restrict__ fcW_bf,
    const float* __restrict__ fc_b, bfbits* __restrict__ h0)
{
    int wave = blockIdx.x * (blockDim.x >> 5) + (threadIdx.x >> 5);
    int lane = threadIdx.x & 31;
    if (wave >= (BATCH / 16) * (HID / 16)) return;
    int bt = wave >> 6, jt = wave & 63;
    int b0 = bt * 16, j0 = jt * 16;

    v8f acc = vzero8();
    for (int k = 0; k < LATENT; k += 32) {
        v16bf a = load_a_frag(lat_bf, LATENT, b0, k, lane);
        v16bf b = load_b_frag(fcW_bf, LATENT, j0, k, lane);
        acc = WMMA_BF16(a, b, acc);
    }
    int n = lane & 15, mh = (lane >> 4) << 3;
    float bb = fc_b[j0 + n];
#pragma unroll
    for (int v = 0; v < 8; ++v) {
        int row = b0 + mh + v;
        h0[(size_t)row * HID + j0 + n] = f2bf(acc[v] + bb);
    }
}

// ---------------------------------------------------------------- output projection tile (shared)
// y_t = h_t @ out_W^T + out_b ; one wave per 16x16 tile of [BATCH, OUTD].
// Mapping: jt = w>>4, bt = w&15 so the 8 waves of a block share the same
// out_W j-tile (B fragments hit in WGP$ after the first wave).
__device__ __forceinline__ void proj_tile(
    const bfbits* __restrict__ h, const bfbits* __restrict__ outW_bf,
    const float* __restrict__ out_b, float* __restrict__ y, int t, int w, int lane)
{
    int jt = w >> 4, bt = w & 15;
    int b0 = bt * 16, j0 = jt * 16;
    v8f acc = vzero8();
    for (int k = 0; k < HID; k += 32) {
        v16bf a = load_a_frag(h, HID, b0, k, lane);
        v16bf b = load_b_frag(outW_bf, HID, j0, k, lane);
        acc = WMMA_BF16(a, b, acc);
    }
    int n = lane & 15, mh = (lane >> 4) << 3;
    float ob = out_b[j0 + n];
#pragma unroll
    for (int v = 0; v < 8; ++v) {
        int row = b0 + mh + v;
        y[(size_t)row * SEQLEN * OUTD + (size_t)t * OUTD + j0 + n] = acc[v] + ob;
    }
}

// ---------------------------------------------------------------- kernel 4: fused LSTM step
// blocks 0..63   : h_t, c_t update. One j-tile per block; the block's 8 waves
//                  cover all 256 batch rows (32 rows each, register-blocked
//                  M=32 -> 2 WMMAs per B fragment; B shared across waves).
// blocks 64..95  : y_{t-1} = h_{t-1} @ out_W^T + out_b  (if do_y)
__global__ __launch_bounds__(256) void step_kernel(
    const bfbits* __restrict__ h_in, bfbits* __restrict__ h_out,
    float* __restrict__ cstate,
    const bfbits* __restrict__ Wg, const float* __restrict__ bias,
    const bfbits* __restrict__ outW_bf, const float* __restrict__ out_b,
    float* __restrict__ y, int do_y, int y_t)
{
    int wid  = threadIdx.x >> 5;
    int lane = threadIdx.x & 31;

    if (blockIdx.x < HID / 16) {
        int j0 = blockIdx.x * 16;          // this block's hidden-column tile
        int b0 = wid * 32;                 // 8 waves x 32 rows = 256 = BATCH

        v8f acc[4][2];
#pragma unroll
        for (int g = 0; g < 4; ++g) {
            acc[g][0] = vzero8();
            acc[g][1] = vzero8();
        }

        for (int k = 0; k < HID; k += 32) {
            v16bf a0 = load_a_frag(h_in, HID, b0,      k, lane);
            v16bf a1 = load_a_frag(h_in, HID, b0 + 16, k, lane);
#pragma unroll
            for (int g = 0; g < 4; ++g) {
                v16bf b = load_b_frag(Wg, HID, g * HID + j0, k, lane);
                acc[g][0] = WMMA_BF16(a0, b, acc[g][0]);
                acc[g][1] = WMMA_BF16(a1, b, acc[g][1]);
            }
        }

        int n = lane & 15, mh = (lane >> 4) << 3;
        int col = j0 + n;
        float bi = bias[0 * HID + col];
        float bf = bias[1 * HID + col];
        float bg = bias[2 * HID + col];
        float bo = bias[3 * HID + col];
#pragma unroll
        for (int mt = 0; mt < 2; ++mt) {
#pragma unroll
            for (int v = 0; v < 8; ++v) {
                int row = b0 + mt * 16 + mh + v;
                size_t idx = (size_t)row * HID + col;
                float gi = sigmoidf_(acc[0][mt][v] + bi);
                float gf = sigmoidf_(acc[1][mt][v] + bf);
                float gg = tanhf(acc[2][mt][v] + bg);
                float go = sigmoidf_(acc[3][mt][v] + bo);
                float cc = gf * cstate[idx] + gi * gg;
                cstate[idx] = cc;
                h_out[idx] = f2bf(go * tanhf(cc));
            }
        }
    } else if (do_y) {
        int w = (blockIdx.x - HID / 16) * 8 + wid;   // 0..255
        if (w < (BATCH / 16) * (OUTD / 16))
            proj_tile(h_in, outW_bf, out_b, y, y_t, w, lane);
    }
}

// ---------------------------------------------------------------- kernel 5: final y projection
__global__ __launch_bounds__(256) void proj_kernel(
    const bfbits* __restrict__ h, const bfbits* __restrict__ outW_bf,
    const float* __restrict__ out_b, float* __restrict__ y, int t)
{
    int wave = blockIdx.x * (blockDim.x >> 5) + (threadIdx.x >> 5);
    int lane = threadIdx.x & 31;
    if (wave >= (BATCH / 16) * (OUTD / 16)) return;
    proj_tile(h, outW_bf, out_b, y, t, wave, lane);
}

// ---------------------------------------------------------------- host launcher
extern "C" void kernel_launch(void* const* d_in, const int* in_sizes, int n_in,
                              void* d_out, int out_size, void* d_ws, size_t ws_size,
                              hipStream_t stream)
{
    const float* latent = (const float*)d_in[0];
    const float* fc_W   = (const float*)d_in[1];
    const float* fc_b   = (const float*)d_in[2];
    const float* W_ih   = (const float*)d_in[3];
    const float* W_hh   = (const float*)d_in[4];
    const float* b_ih   = (const float*)d_in[5];
    const float* b_hh   = (const float*)d_in[6];
    const float* out_W  = (const float*)d_in[7];
    const float* out_b  = (const float*)d_in[8];
    float* y = (float*)d_out;

    char* ws = (char*)d_ws;
    size_t off = 0;
    auto carve = [&](size_t bytes) -> char* {
        char* p = ws + off;
        off += (bytes + 255) & ~(size_t)255;
        return p;
    };
    bfbits* Weff     = (bfbits*)carve((size_t)4 * HID * HID * 2);     // 8 MB
    bfbits* Whh_bf   = (bfbits*)carve((size_t)4 * HID * HID * 2);     // 8 MB
    bfbits* Wih_bf   = (bfbits*)carve((size_t)4 * HID * OUTD * 2);    // 2 MB
    bfbits* outW_bf  = (bfbits*)carve((size_t)OUTD * HID * 2);
    bfbits* outWT_bf = (bfbits*)carve((size_t)HID * OUTD * 2);
    bfbits* fcW_bf   = (bfbits*)carve((size_t)HID * LATENT * 2);
    bfbits* lat_bf   = (bfbits*)carve((size_t)BATCH * LATENT * 2);
    bfbits* hbuf0    = (bfbits*)carve((size_t)BATCH * HID * 2);
    bfbits* hbuf1    = (bfbits*)carve((size_t)BATCH * HID * 2);
    float*  cstate   = (float*) carve((size_t)BATCH * HID * 4);
    float*  bias0    = (float*) carve((size_t)4 * HID * 4);
    float*  bias_eff = (float*) carve((size_t)4 * HID * 4);
    (void)ws_size; (void)in_sizes; (void)n_in; (void)out_size;

    // 1) precision conversion, transposes, fused biases, c0 = 0
    convert_kernel<<<2048, 256, 0, stream>>>(
        latent, fc_W, W_ih, W_hh, b_ih, b_hh, out_W, out_b,
        lat_bf, fcW_bf, Wih_bf, Whh_bf, outW_bf, outWT_bf,
        bias0, bias_eff, cstate);

    // 2) W_eff = W_hh + W_ih @ out_W   (folds output projection into recurrence)
    weff_kernel<<<2048, 256, 0, stream>>>(Wih_bf, outWT_bf, W_hh, Weff);

    // 3) h0 = latent @ fc_W^T + fc_b
    h0_kernel<<<128, 256, 0, stream>>>(lat_bf, fcW_bf, fc_b, hbuf0);

    // 4) 128 fused steps: h_t/c_t update + y_{t-1} projection in one launch
    for (int t = 0; t < SEQLEN; ++t) {
        const bfbits* hin  = (t & 1) ? hbuf1 : hbuf0;
        bfbits*       hout = (t & 1) ? hbuf0 : hbuf1;
        const bfbits* Wg   = t ? Weff : Whh_bf;          // t=0: x0 == 0, no folded term
        const float*  bs   = t ? bias_eff : bias0;
        step_kernel<<<96, 256, 0, stream>>>(hin, hout, cstate, Wg, bs,
                                            outW_bf, out_b, y, t > 0 ? 1 : 0, t - 1);
    }

    // 5) final output row y_{SEQ-1} from h_{127} (lives in hbuf0: 128 is even)
    proj_kernel<<<32, 256, 0, stream>>>(hbuf0, outW_bf, out_b, y, SEQLEN - 1);
}
